// MultiHeadAttention_35682588295616
// MI455X (gfx1250) — matrix operand order repaired and checked by reading, not compiled, not run on hardware
//
#include <hip/hip_runtime.h>
#include <hip/hip_bf16.h>

typedef __bf16 bf16;
typedef __attribute__((ext_vector_type(8)))  bf16  v8bf;
typedef __attribute__((ext_vector_type(16))) bf16  v16bf;
typedef __attribute__((ext_vector_type(8)))  float v8f;
typedef __attribute__((ext_vector_type(4)))  unsigned int v4u;
typedef __attribute__((ext_vector_type(8)))  int  v8i;
typedef __attribute__((ext_vector_type(4)))  int  v4i;

#define B_  2
#define S_  2048
#define D_  1024
#define H_  16
#define DH_ 64

// LDS A-tile row stride in elements: 32 payload + 8 pad (16B pad per 64B row,
// inserted by the TDM pad feature) -> 20-bank stride, conflict-free b128 reads.
#define LDS_AW 40

// ---------------------------------------------------------------------------
// WMMA fragment helpers (layouts per CDNA5 ISA 7.12.2, bf16 16x16x32)
// ---------------------------------------------------------------------------

static __device__ __forceinline__ v8f wmma_bf16(v16bf a, v16bf b, v8f c) {
  return __builtin_amdgcn_wmma_f32_16x16x32_bf16(false, a, false, b, (short)0, c,
                                                 false, false);
}

// A fragment, 16x32 (MxK), source row-major with leading dim `ld` (elements).
// lane(mrow = lane&15, half = lane>>4):
//   elements 0..7  <- K = half*8 + 0..7
//   elements 8..15 <- K = half*8 + 16..23
static __device__ __forceinline__ v16bf load_a_frag(const bf16* base,
                                                    int ld, int mrow, int half) {
  const bf16* p = base + (size_t)mrow * ld + half * 8;
  v8bf lo = *reinterpret_cast<const v8bf*>(p);
  v8bf hi = *reinterpret_cast<const v8bf*>(p + 16);
  v16bf a;
#pragma unroll
  for (int i = 0; i < 8; ++i) { a[i] = lo[i]; a[i + 8] = hi[i]; }
  return a;
}

// B fragment, 32x16 (KxN), source stored N-major (Bt[n][k], leading dim over k).
// lane holds column N = lane&15; elements 0..7 <- K = half*8 + 0..7, 8..15 <- K = half*8 + 16..23.
static __device__ __forceinline__ v16bf load_b_frag(const bf16* __restrict__ baseT,
                                                    int ld, int ncol, int half) {
  const bf16* p = baseT + (size_t)ncol * ld + half * 8;
  v8bf lo = *reinterpret_cast<const v8bf*>(p);
  v8bf hi = *reinterpret_cast<const v8bf*>(p + 16);
  v16bf b;
#pragma unroll
  for (int i = 0; i < 8; ++i) { b[i] = lo[i]; b[i + 8] = hi[i]; }
  return b;
}

// ---------------------------------------------------------------------------
// Tensor Data Mover: 2D tile (tile_w x tile_h elements, bf16) from a row-major
// [4096 x 1024] bf16 tensor into LDS, with 16B LDS padding after every 64B row
// (pad_interval code 3 = 16 DWORDs, pad_amount code 3 = 4 DWORDs).
// Descriptor layout per CDNA5 ISA 08_async_tensor §8.3-8.6.
// This toolchain exposes the 6-arg builtin:
//   (uint32x4 g0, int32x8 g1, int32x4 g2, int32x4 g3, int32x8 g4, i32 cpol)
// ---------------------------------------------------------------------------
static __device__ __forceinline__ void tdm_load_a(unsigned lds_off,
                                                  const bf16* gsrc,
                                                  int tile_w, int tile_h) {
  const unsigned long long ga = (unsigned long long)(size_t)gsrc;
  v4u g0;
  g0[0] = 1u;                                   // count=1 (valid user descriptor)
  g0[1] = lds_off;                              // lds_addr (bytes)
  g0[2] = (unsigned)(ga & 0xffffffffull);       // global_addr[31:0]
  g0[3] = (unsigned)((ga >> 32) & 0x1ffffffull) // global_addr[56:32]
          | (2u << 30);                         // type = 2 ("image")
  v8i g1;
  g1[0] = (1 << 16)                             // data_size = 2 bytes
        | (1 << 20)                             // pad_enable
        | (3 << 22)                             // pad_interval: 16 DWORDs (64B)
        | (3 << 25);                            // pad_amount: 4 DWORDs (16B)
  g1[1] = (D_ & 0xffff) << 16;                  // tensor_dim0[15:0] (=1024)
  g1[2] = (D_ >> 16) | (((B_ * S_) & 0xffff) << 16);  // td0 hi | tensor_dim1 lo
  g1[3] = ((B_ * S_) >> 16) | (tile_w << 16);   // td1 hi | tile_dim0
  g1[4] = tile_h & 0xffff;                      // tile_dim1 (tile_dim2 = 0)
  g1[5] = D_;                                   // tensor_dim0_stride[31:0]
  g1[6] = 0;                                    // stride hi | td1_stride lo
  g1[7] = 0;
  v4i g2 = (v4i){0, 0, 0, 0};                   // 2D: dims 2..4 unused
  v4i g3 = (v4i){0, 0, 0, 0};
  v8i g4 = (v8i){0, 0, 0, 0, 0, 0, 0, 0};      // unused trailing group
  __builtin_amdgcn_tensor_load_to_lds(g0, g1, g2, g3, g4, 0);
}

// ---------------------------------------------------------------------------
// fp32 -> bf16 convert / transpose
// ---------------------------------------------------------------------------

__global__ __launch_bounds__(256) void cvt_kernel(const float* __restrict__ in,
                                                  bf16* __restrict__ out, int n) {
  const int i = blockIdx.x * 256 + threadIdx.x;
  if (i < n) out[i] = (bf16)in[i];
}

// out[n][k] = (bf16) in[k][n]   (square D_ x D_)
__global__ __launch_bounds__(256) void transpose_kernel(const float* __restrict__ in,
                                                        bf16* __restrict__ out) {
  const int n = blockIdx.x * 256 + threadIdx.x;
  const int k = blockIdx.y;
  out[(size_t)n * D_ + k] = (bf16)in[(size_t)k * D_ + n];
}

// ---------------------------------------------------------------------------
// GEMM: C[4096 x 1024] = X[4096 x 1024] * W + bias
//   X bf16 row-major, W given transposed (Wt[n][k] bf16).
//   mode 0: out bf16, head layout [B,H,S,DH]    (Q / K projections)
//   mode 1: out bf16, transposed  [B,H,DH,S]    (V projection)
//   mode 2: out fp32, plain [4096,1024]         (output projection)
// Block = 8 waves sharing one 64-row A tile; TDM double-buffers the A tile
// (64 x 32 bf16) into LDS; each wave computes a 64x32 tile (8 WMMA / K-step).
// ---------------------------------------------------------------------------

__global__ __launch_bounds__(256) void wmma_gemm(const bf16* __restrict__ X,
                                                 const bf16* __restrict__ Wt,
                                                 const float* __restrict__ bias,
                                                 void* __restrict__ out,
                                                 int mode, float scale) {
  __shared__ __align__(16) bf16 ldsA[2][64 * LDS_AW];

  const int lane = threadIdx.x & 31;
  const int wave = threadIdx.x >> 5;
  const int half = lane >> 4;
  const int lrow = lane & 15;  // row for A, column for B/C

  const int n0 = (blockIdx.x * 8 + wave) * 32;
  const int m0 = blockIdx.y * 64;

  const unsigned ldsOff0 = (unsigned)(size_t)(&ldsA[0][0]);
  const unsigned bufBytes = 64 * LDS_AW * 2;

  v8f acc[4][2];
#pragma unroll
  for (int mi = 0; mi < 4; ++mi)
#pragma unroll
    for (int ni = 0; ni < 2; ++ni)
      acc[mi][ni] = (v8f){0.f, 0.f, 0.f, 0.f, 0.f, 0.f, 0.f, 0.f};

  // prologue: stage first A tile
  if (wave == 0)
    tdm_load_a(ldsOff0, X + (size_t)m0 * D_, 32, 64);

  int cur = 0;
  for (int kk = 0; kk < D_; kk += 32, cur ^= 1) {
    if (wave == 0) {
      if (kk + 32 < D_) {
        tdm_load_a(ldsOff0 + (cur ^ 1) * bufBytes,
                   X + (size_t)m0 * D_ + kk + 32, 32, 64);
        __builtin_amdgcn_s_wait_tensorcnt(1);  // older (current) TDM done
      } else {
        __builtin_amdgcn_s_wait_tensorcnt(0);
      }
    }
    __syncthreads();  // A tile [cur] visible to all waves

    v16bf af[4];
#pragma unroll
    for (int mi = 0; mi < 4; ++mi)
      af[mi] = load_a_frag(&ldsA[cur][(size_t)(mi * 16) * LDS_AW],
                           LDS_AW, lrow, half);
    v16bf bfr[2];
#pragma unroll
    for (int ni = 0; ni < 2; ++ni)
      bfr[ni] = load_b_frag(Wt + (size_t)(n0 + ni * 16) * D_ + kk, D_, lrow, half);
#pragma unroll
    for (int mi = 0; mi < 4; ++mi)
#pragma unroll
      for (int ni = 0; ni < 2; ++ni)
        acc[mi][ni] = wmma_bf16(af[mi], bfr[ni], acc[mi][ni]);

    __syncthreads();  // everyone done reading [cur] before it is overwritten
  }

#pragma unroll
  for (int ni = 0; ni < 2; ++ni) {
    const int n = n0 + ni * 16 + lrow;
    const float bv = bias[n];
#pragma unroll
    for (int mi = 0; mi < 4; ++mi) {
#pragma unroll
      for (int r = 0; r < 8; ++r) {
        const int m = m0 + mi * 16 + half * 8 + r;
        const float v = (acc[mi][ni][r] + bv) * scale;
        if (mode == 2) {
          ((float*)out)[(size_t)m * D_ + n] = v;
        } else {
          const int b = m >> 11, s = m & (S_ - 1);
          const int h = n >> 6, d = n & (DH_ - 1);
          if (mode == 0)
            ((bf16*)out)[(((size_t)(b * H_ + h) * S_) + s) * DH_ + d] = (bf16)v;
          else  // V transposed: [B,H,DH,S]
            ((bf16*)out)[(((size_t)(b * H_ + h) * DH_) + d) * S_ + s] = (bf16)v;
        }
      }
    }
  }
}

// ---------------------------------------------------------------------------
// Flash attention: one wave per 16-query tile of one (b,h).
//   Qh [B,H,S,DH] (pre-scaled by 1/sqrt(DH)), Kh [B,H,S,DH], Vt [B,H,DH,S].
//   k-loop in blocks of 32 keys: 4 score WMMAs + online softmax + 4 ctx WMMAs.
// ---------------------------------------------------------------------------

__global__ __launch_bounds__(256) void attn_kernel(const bf16* __restrict__ Qh,
                                                   const bf16* __restrict__ Kh,
                                                   const bf16* __restrict__ Vt,
                                                   bf16* __restrict__ ctx) {
  __shared__ __align__(16) bf16 ldsP[8][16][48];

  const int lane = threadIdx.x & 31;
  const int wave = threadIdx.x >> 5;
  const int half = lane >> 4;
  const int lrow = lane & 15;

  const int bh = blockIdx.x;  // b*H + h
  const int b = bh >> 4, h = bh & 15;
  const int q0 = (blockIdx.y * 8 + wave) * 16;

  const bf16* Qp = Qh + ((size_t)bh * S_ + q0) * DH_;
  const bf16* Kp = Kh + (size_t)bh * S_ * DH_;
  const bf16* Vp = Vt + (size_t)bh * DH_ * S_;

  v16bf aq[2];
#pragma unroll
  for (int kd = 0; kd < 2; ++kd) aq[kd] = load_a_frag(Qp + kd * 32, DH_, lrow, half);

  float rmax[8], lsum[8];
#pragma unroll
  for (int r = 0; r < 8; ++r) { rmax[r] = -1e30f; lsum[r] = 0.f; }
  v8f acc[4];
#pragma unroll
  for (int di = 0; di < 4; ++di)
    acc[di] = (v8f){0.f, 0.f, 0.f, 0.f, 0.f, 0.f, 0.f, 0.f};

  for (int k0 = 0; k0 < S_; k0 += 32) {
    v8f c[2];
#pragma unroll
    for (int ni = 0; ni < 2; ++ni) {
      c[ni] = (v8f){0.f, 0.f, 0.f, 0.f, 0.f, 0.f, 0.f, 0.f};
#pragma unroll
      for (int kd = 0; kd < 2; ++kd) {
        v16bf kb = load_b_frag(Kp + (size_t)(k0 + ni * 16) * DH_ + kd * 32,
                               DH_, lrow, half);
        c[ni] = wmma_bf16(aq[kd], kb, c[ni]);
      }
    }

    // online softmax; row r of this half lives across lanes of the half-wave
#pragma unroll
    for (int r = 0; r < 8; ++r) {
      float s0 = c[0][r], s1 = c[1][r];
      float mx = fmaxf(s0, s1);
#pragma unroll
      for (int off = 8; off >= 1; off >>= 1) mx = fmaxf(mx, __shfl_xor(mx, off, 32));
      const float mnew = fmaxf(rmax[r], mx);
      const float alpha = __expf(rmax[r] - mnew);
      rmax[r] = mnew;
      const float p0 = __expf(s0 - mnew);
      const float p1 = __expf(s1 - mnew);
      float rs = p0 + p1;
#pragma unroll
      for (int off = 8; off >= 1; off >>= 1) rs += __shfl_xor(rs, off, 32);
      lsum[r] = lsum[r] * alpha + rs;
#pragma unroll
      for (int di = 0; di < 4; ++di) acc[di][r] = acc[di][r] * alpha;
      ldsP[wave][half * 8 + r][lrow] = (bf16)p0;
      ldsP[wave][half * 8 + r][16 + lrow] = (bf16)p1;
    }

    // same-wave LDS RAW: LDS is in-order; force scheduler ordering + drain
    asm volatile("s_wait_dscnt 0" ::: "memory");

    v16bf pf;
    {
      const bf16* p = &ldsP[wave][lrow][half * 8];
      v8bf lo = *reinterpret_cast<const v8bf*>(p);
      v8bf hi = *reinterpret_cast<const v8bf*>(p + 16);
#pragma unroll
      for (int i = 0; i < 8; ++i) { pf[i] = lo[i]; pf[i + 8] = hi[i]; }
    }

#pragma unroll
    for (int di = 0; di < 4; ++di) {
      v16bf vb = load_b_frag(Vp + (size_t)(di * 16) * S_ + k0, S_, lrow, half);
      acc[di] = wmma_bf16(pf, vb, acc[di]);
    }
  }

#pragma unroll
  for (int di = 0; di < 4; ++di) {
#pragma unroll
    for (int r = 0; r < 8; ++r) {
      const int row = half * 8 + r;
      const float v = acc[di][r] / lsum[r];
      ctx[((size_t)(b * S_ + q0 + row)) * D_ + h * DH_ + di * 16 + lrow] = (bf16)v;
    }
  }
}

// ---------------------------------------------------------------------------
// Host-side launch
// ---------------------------------------------------------------------------

extern "C" void kernel_launch(void* const* d_in, const int* in_sizes, int n_in,
                              void* d_out, int out_size, void* d_ws, size_t ws_size,
                              hipStream_t stream) {
  (void)in_sizes; (void)n_in; (void)out_size; (void)ws_size;

  const float* q  = (const float*)d_in[0];
  const float* k  = (const float*)d_in[1];
  const float* v  = (const float*)d_in[2];
  const float* wq = (const float*)d_in[3];
  const float* bq = (const float*)d_in[4];
  const float* wk = (const float*)d_in[5];
  const float* bk = (const float*)d_in[6];
  const float* wv = (const float*)d_in[7];
  const float* bv = (const float*)d_in[8];
  const float* wo = (const float*)d_in[9];
  const float* bo = (const float*)d_in[10];

  // workspace layout (bf16 elements): 7*4Mi + 4*1Mi = 32Mi elems = 64 MiB
  const size_t TOK = (size_t)B_ * S_ * D_;  // 4,194,304
  const size_t WEL = (size_t)D_ * D_;       // 1,048,576
  bf16* base = (bf16*)d_ws;
  bf16* xq  = base;
  bf16* xk  = xq  + TOK;
  bf16* xv  = xk  + TOK;
  bf16* wqt = xv  + TOK;
  bf16* wkt = wqt + WEL;
  bf16* wvt = wkt + WEL;
  bf16* wot = wvt + WEL;
  bf16* Qh  = wot + WEL;
  bf16* Kh  = Qh  + TOK;
  bf16* Vt  = Kh  + TOK;
  bf16* ctx = Vt  + TOK;

  const dim3 cb(256);

  cvt_kernel<<<dim3((unsigned)(TOK / 256)), cb, 0, stream>>>(q, xq, (int)TOK);
  cvt_kernel<<<dim3((unsigned)(TOK / 256)), cb, 0, stream>>>(k, xk, (int)TOK);
  cvt_kernel<<<dim3((unsigned)(TOK / 256)), cb, 0, stream>>>(v, xv, (int)TOK);
  transpose_kernel<<<dim3(D_ / 256, D_), cb, 0, stream>>>(wq, wqt);
  transpose_kernel<<<dim3(D_ / 256, D_), cb, 0, stream>>>(wk, wkt);
  transpose_kernel<<<dim3(D_ / 256, D_), cb, 0, stream>>>(wv, wvt);
  transpose_kernel<<<dim3(D_ / 256, D_), cb, 0, stream>>>(wo, wot);

  const dim3 gg(4, (B_ * S_) / 64);
  wmma_gemm<<<gg, cb, 0, stream>>>(xq, wqt, bq, (void*)Qh, 0, 0.125f);  // 1/sqrt(64)
  wmma_gemm<<<gg, cb, 0, stream>>>(xk, wkt, bk, (void*)Kh, 0, 1.0f);
  wmma_gemm<<<gg, cb, 0, stream>>>(xv, wvt, bv, (void*)Vt, 1, 1.0f);

  attn_kernel<<<dim3(B_ * H_, S_ / (16 * 8)), cb, 0, stream>>>(Qh, Kh, Vt, ctx);

  wmma_gemm<<<gg, cb, 0, stream>>>(ctx, wot, bo, d_out, 2, 1.0f);
}
